// TransformerBlock_30683246362953
// MI455X (gfx1250) — matrix operand order repaired and checked by reading, not compiled, not run on hardware
//
#include <hip/hip_runtime.h>
#include <hip/hip_bf16.h>
#include <math.h>

#define DIMC 192
#define HEADS 6
#define HDIM 32
#define TIMED 128
#define HID 510
#define HID2 1020
#define REDC 24
#define BATCH 4
#define HH 256
#define WW 256
#define HW (HH*WW)          // 65536
#define NPIX (BATCH*HW)     // 262144

typedef __attribute__((ext_vector_type(16))) _Float16 v16h;
typedef __attribute__((ext_vector_type(8)))  _Float16 v8h;
typedef __attribute__((ext_vector_type(2)))  _Float16 v2h;
typedef __attribute__((ext_vector_type(8)))  float    v8f;

// ---------------- WMMA fragment helpers (wave32, A and B both use contiguous k = 16*half + j) ----------------

__device__ __forceinline__ v8f wmma_f16(v16h a, v16h b, v8f c) {
    return __builtin_amdgcn_wmma_f32_16x16x32_f16(false, a, false, b, (short)0, c, false, false);
}

// A fragment from row-major src[16][ld], cols kbase..+31.
// Per lane the needed halves are two contiguous 16B runs:
//   elements 0..7  <- K = kbase + half*16     .. +7   (VGPRs 0..3)
//   elements 8..15 <- K = kbase + half*16 + 8 .. +7   (VGPRs 4..7)
__device__ __forceinline__ v16h load_A16x32(const _Float16* __restrict__ src, int ld, int kbase) {
    int lane = threadIdx.x & 31;
    int m = lane & 15, half = lane >> 4;
    const _Float16* p = src + m*ld + kbase + half*16;
    v8h lo = *(const v8h*)p;
    v8h hi = *(const v8h*)(p + 8);
    return __builtin_shufflevector(lo, hi, 0,1,2,3,4,5,6,7,8,9,10,11,12,13,14,15);
}

// B fragment from fragment-major packed weights: one aligned 32B load per lane.
// Packed layout: frag blocks of 512 halves; half j of lane l is
//   k = ktile*32 + (l>>4)*16 + j ,  n = ntile*16 + (l&15)
__device__ __forceinline__ v16h load_Bfrag(const _Float16* __restrict__ packed, int frag) {
    return *(const v16h*)(packed + ((size_t)frag << 9) + ((threadIdx.x & 31) << 4));
}

// B fragment from an n-major source (row per N, contiguous K): B[k][n] = src[n*ld + k].
// Per lane: 16 contiguous halves at src[n*ld + half*16] -> single 32B load.
__device__ __forceinline__ v16h load_BfromRows(const _Float16* __restrict__ src, int ld) {
    int lane = threadIdx.x & 31;
    int n = lane & 15, half = lane >> 4;
    return *(const v16h*)(src + n*ld + half*16);
}

// ---------------- weight packing kernels ----------------

// Generic B packer: dst frag = ntile*(Kp/32) + ktile, 512 halves each.
// mode 0: val = src[k*ld + n] ; mode 1: val = src[n*ld + k] ; zero outside valid range.
__global__ void packB_kernel(const float* __restrict__ src, _Float16* __restrict__ dst,
                             int Kp, int Np, int kvalid, int nvalid, int ld, int mode) {
    int i = blockIdx.x*blockDim.x + threadIdx.x;
    if (i >= Kp*Np) return;
    int frag = i >> 9, r = i & 511;
    int lane = r >> 4, j = r & 15;
    int KT = Kp >> 5;
    int nt = frag / KT, kt = frag % KT;
    int k = kt*32 + (lane >> 4)*16 + j;
    int n = nt*16 + (lane & 15);
    float v = 0.f;
    if (k < kvalid && n < nvalid)
        v = (mode == 0) ? src[(size_t)k*ld + n] : src[(size_t)n*ld + k];
    dst[i] = (_Float16)v;
}

// dw_w (1020,1,3,3) -> dwwT[9][1024] so channel-consecutive threads load coalesced
__global__ void dwwT_kernel(const float* __restrict__ dww, float* __restrict__ dst) {
    int i = blockIdx.x*blockDim.x + threadIdx.x;
    if (i >= 9*1024) return;
    int w = i >> 10, c = i & 1023;
    dst[i] = (c < HID2) ? dww[c*9 + w] : 0.f;
}

// ---------------- reductions / tiny MLPs ----------------

__global__ __launch_bounds__(256) void moments_kernel(const float* __restrict__ x,
                                                      float* __restrict__ stats,
                                                      int perSample, int blocksPerSample) {
    int sample = blockIdx.x / blocksPerSample;
    int blk    = blockIdx.x % blocksPerSample;
    const float4* p = (const float4*)(x + (size_t)sample*perSample);
    int n4 = perSample >> 2;
    float s = 0.f, ss = 0.f;
    for (int i = blk*blockDim.x + threadIdx.x; i < n4; i += blocksPerSample*blockDim.x) {
        float4 v = p[i];
        s  += v.x + v.y + v.z + v.w;
        ss += v.x*v.x + v.y*v.y + v.z*v.z + v.w*v.w;
    }
    __shared__ float rs[256], rq[256];
    rs[threadIdx.x] = s; rq[threadIdx.x] = ss; __syncthreads();
    for (int o = 128; o > 0; o >>= 1) {
        if (threadIdx.x < o) { rs[threadIdx.x] += rs[threadIdx.x+o]; rq[threadIdx.x] += rq[threadIdx.x+o]; }
        __syncthreads();
    }
    if (threadIdx.x == 0) {
        atomicAdd(&stats[sample*2+0], rs[0]);
        atomicAdd(&stats[sample*2+1], rq[0]);
    }
}

__global__ __launch_bounds__(TIMED) void scale_mlp_kernel(const float* __restrict__ temb,
                                                          const float* __restrict__ w1,
                                                          const float* __restrict__ b1,
                                                          const float* __restrict__ w2,
                                                          const float* __restrict__ b2,
                                                          float* __restrict__ s) {
    int b = blockIdx.x, t = threadIdx.x;
    __shared__ float hid[TIMED];
    const float* te = temb + b*TIMED;
    float a = b1[t];
    for (int i = 0; i < TIMED; ++i) a += te[i]*w1[i*TIMED + t];
    hid[t] = a / (1.f + expf(-a));   // silu
    __syncthreads();
    for (int o = t; o < 4*DIMC; o += TIMED) {
        float acc = b2[o];
        for (int j = 0; j < TIMED; ++j) acc += hid[j]*w2[j*(4*DIMC) + o];
        s[b*(4*DIMC) + o] = acc;
    }
}

// ---------------- fused GN + Haar DWT*s + IDWT, channel-pair vectorized ----------------
// NCHW f32 -> NHWC f16
__global__ void norm_adawm_kernel(const float* __restrict__ x, const float* __restrict__ stats,
                                  const float* __restrict__ gamma, const float* __restrict__ beta,
                                  const float* __restrict__ s, _Float16* __restrict__ t) {
    size_t idx = (size_t)blockIdx.x*blockDim.x + threadIdx.x;
    if (idx >= (size_t)BATCH*(DIMC/2)*16384) return;
    int xh = (int)(idx & 127);
    int yh = (int)((idx >> 7) & 127);
    int cp = (int)((idx >> 14) % (DIMC/2));
    int b  = (int)(idx / ((size_t)(DIMC/2)*16384));
    int c0 = cp*2;
    const float N = (float)(DIMC*HW);
    float mu  = stats[b*2+0] / N;
    float var = stats[b*2+1] / N - mu*mu;
    float inv = rsqrtf(var + 1e-5f);
    int y = 2*yh, xx = 2*xh;
    const float* sb = s + b*(4*DIMC);
    v2h r00, r01, r10, r11;
#pragma unroll
    for (int q = 0; q < 2; ++q) {
        int c = c0 + q;
        float ga = gamma[c]*inv;
        float be = beta[c] - mu*ga;
        const float* xp = x + ((size_t)(b*DIMC + c))*HW;
        float2 top = *(const float2*)(xp + (size_t)y*WW + xx);
        float2 bot = *(const float2*)(xp + (size_t)(y+1)*WW + xx);
        float v00 = top.x*ga + be, v01 = top.y*ga + be;
        float v10 = bot.x*ga + be, v11 = bot.y*ga + be;
        float ll = (v00 + v01 + v10 + v11)*0.5f*sb[c];
        float hl = (v00 - v10 + v01 - v11)*0.5f*sb[DIMC   + c];
        float lh = (v00 + v10 - v01 - v11)*0.5f*sb[2*DIMC + c];
        float hh = (v00 - v10 - v01 + v11)*0.5f*sb[3*DIMC + c];
        r00[q] = (_Float16)((ll + hl + lh + hh)*0.5f);
        r01[q] = (_Float16)((ll + hl - lh - hh)*0.5f);
        r10[q] = (_Float16)((ll - hl + lh - hh)*0.5f);
        r11[q] = (_Float16)((ll - hl - lh + hh)*0.5f);
    }
    size_t base = ((size_t)((b*HH + y)*WW + xx))*DIMC + c0;
    *(v2h*)(t + base)                          = r00;
    *(v2h*)(t + base + DIMC)                   = r01;
    *(v2h*)(t + base + (size_t)WW*DIMC)        = r10;
    *(v2h*)(t + base + (size_t)WW*DIMC + DIMC) = r11;
}

// ---------------- window attention: one wave per 4x4 window ----------------

__global__ __launch_bounds__(32) void win_attn_kernel(const _Float16* __restrict__ t1,   // NHWC f16
                                                      const float*    __restrict__ x_in, // NCHW f32
                                                      const _Float16* __restrict__ qkvP, // packed frags
                                                      const float*    __restrict__ qkvb,
                                                      const _Float16* __restrict__ projP,// packed frags
                                                      const float*    __restrict__ projb,
                                                      float*          __restrict__ x2) { // NCHW f32
    __shared__ __align__(32) _Float16 Xs[16*DIMC];
    __shared__ __align__(32) _Float16 QKVs[16*3*DIMC];
    __shared__ __align__(32) float    Ss[16*16];
    __shared__ __align__(32) _Float16 As[16*32];
    __shared__ __align__(32) _Float16 Vt[32*32];
    __shared__ __align__(32) _Float16 Os[16*DIMC];

    const int lane = threadIdx.x;
    const int win = blockIdx.x;
    const int bx = win & 63, by = (win >> 6) & 63, b = win >> 12;
    const int y0 = by*4, x0 = bx*4;
    const int n = lane & 15, half = lane >> 4;

    // stage 16 tokens: contiguous 384B rows (NHWC), copied as uint2
    for (int t = 0; t < 16; ++t) {
        const uint2* src = (const uint2*)(t1 + ((size_t)((b*HH + y0 + (t>>2))*WW + x0 + (t&3)))*DIMC);
        uint2* dst = (uint2*)(Xs + t*DIMC);
        for (int c = lane; c < DIMC/4; c += 32) dst[c] = src[c];
    }
    __syncthreads();

    v16h ax[6];
#pragma unroll
    for (int ks = 0; ks < 6; ++ks) ax[ks] = load_A16x32(Xs, DIMC, ks*32);

    // QKV = X(16x192) @ qkv_w(192x576) + b
    for (int nt = 0; nt < 36; ++nt) {
        v8f acc = {};
#pragma unroll
        for (int ks = 0; ks < 6; ++ks)
            acc = wmma_f16(ax[ks], load_Bfrag(qkvP, nt*6 + ks), acc);
        float bias = qkvb[nt*16 + n];
#pragma unroll
        for (int v = 0; v < 8; ++v)
            QKVs[(v + 8*half)*(3*DIMC) + nt*16 + n] = (_Float16)(acc[v] + bias);
    }
    __syncthreads();

    const float sc = 0.17677669529663687f;  // 1/sqrt(HD)
    for (int hd = 0; hd < HEADS; ++hd) {
        // S = Q(16x32) @ K^T : K^T fragment is contiguous in n-major QKVs
        v16h aq = load_A16x32(QKVs, 3*DIMC, hd*HDIM);
        v16h bk = load_BfromRows(QKVs + DIMC + hd*HDIM, 3*DIMC);
        v8f sacc = {};
        sacc = wmma_f16(aq, bk, sacc);
#pragma unroll
        for (int v = 0; v < 8; ++v) Ss[(v + 8*half)*16 + n] = sacc[v]*sc;
        // stage V transposed (32 d-rows x 32 token-cols, tokens>=16 zero)
        for (int i = lane; i < 32*32; i += 32) {
            int d = i >> 5, t = i & 31;
            Vt[i] = (t < 16) ? QKVs[t*(3*DIMC) + 2*DIMC + hd*HDIM + d] : (_Float16)0.f;
        }
        __syncthreads();
        // row softmax on lanes 0..15, probs zero-padded to K=32
        if (lane < 16) {
            float mx = -1e30f;
            for (int k = 0; k < 16; ++k) mx = fmaxf(mx, Ss[lane*16 + k]);
            float e[16]; float sum = 0.f;
            for (int k = 0; k < 16; ++k) { e[k] = __expf(Ss[lane*16 + k] - mx); sum += e[k]; }
            float is = 1.f/sum;
            for (int k = 0; k < 16; ++k) As[lane*32 + k] = (_Float16)(e[k]*is);
            for (int k = 16; k < 32; ++k) As[lane*32 + k] = (_Float16)0.f;
        }
        __syncthreads();
        // O = P(16x16 pad) @ V(16x32): B fragments contiguous from Vt rows
        v16h ap = load_A16x32(As, 32, 0);
#pragma unroll
        for (int part = 0; part < 2; ++part) {
            v16h bv = load_BfromRows(Vt + part*16*32, 32);
            v8f oacc = {};
            oacc = wmma_f16(ap, bv, oacc);
#pragma unroll
            for (int v = 0; v < 8; ++v)
                Os[(v + 8*half)*DIMC + hd*HDIM + part*16 + n] = (_Float16)oacc[v];
        }
        __syncthreads();
    }

    // proj + residual -> x2 (NCHW f32)
    v16h ao[6];
#pragma unroll
    for (int ks = 0; ks < 6; ++ks) ao[ks] = load_A16x32(Os, DIMC, ks*32);
    for (int nt = 0; nt < 12; ++nt) {
        v8f acc = {};
#pragma unroll
        for (int ks = 0; ks < 6; ++ks)
            acc = wmma_f16(ao[ks], load_Bfrag(projP, nt*6 + ks), acc);
        int ch = nt*16 + n;
        float bias = projb[ch];
#pragma unroll
        for (int v = 0; v < 8; ++v) {
            int tok = v + 8*half;
            size_t gi = ((size_t)(b*DIMC + ch))*HW + (size_t)(y0 + (tok>>2))*WW + (x0 + (tok&3));
            x2[gi] = x_in[gi] + acc[v] + bias;
        }
    }
}

// ---------------- GDFN ----------------

// pin: h(16px x 1024ch) = t2(16x192) @ pin (64 ch per wave)
__global__ __launch_bounds__(128) void pin_gemm_kernel(const _Float16* __restrict__ t2,
                                                       const _Float16* __restrict__ pinP,
                                                       const float*    __restrict__ pinb,
                                                       _Float16*       __restrict__ hbuf) {
    int wave = blockIdx.x*(blockDim.x >> 5) + (threadIdx.x >> 5);
    int mt = wave >> 4;
    int ng = wave & 15;
    size_t p0 = (size_t)mt * 16;
    const _Float16* A = t2 + p0*DIMC;
    int n = threadIdx.x & 15, half = (threadIdx.x >> 4) & 1;
    v8f acc[4] = {};
    for (int ks = 0; ks < 6; ++ks) {
        v16h a = load_A16x32(A, DIMC, ks*32);
        if (ks < 5)
            __builtin_prefetch(pinP + ((size_t)((ng*4)*6 + ks + 1) << 9), 0, 3);
#pragma unroll
        for (int nt = 0; nt < 4; ++nt)
            acc[nt] = wmma_f16(a, load_Bfrag(pinP, (ng*4 + nt)*6 + ks), acc[nt]);
    }
#pragma unroll
    for (int nt = 0; nt < 4; ++nt) {
        int ch = ng*64 + nt*16 + n;
        float bias = (ch < HID2) ? pinb[ch] : 0.f;
#pragma unroll
        for (int v = 0; v < 8; ++v) {
            size_t row = p0 + v + 8*half;
            hbuf[row*1024 + ch] = (ch < HID2) ? (_Float16)(acc[nt][v] + bias) : (_Float16)0.f;
        }
    }
}

// depthwise 3x3 (SAME) + exact-GELU gate, channel pairs, coalesced weights
__global__ void dwgate_kernel(const _Float16* __restrict__ hbuf,
                              const float* __restrict__ dwwT,  // [9][1024]
                              const float* __restrict__ dwb,   // [1020]
                              _Float16* __restrict__ g) {
    size_t idx = (size_t)blockIdx.x*blockDim.x + threadIdx.x;
    if (idx >= (size_t)NPIX*256) return;
    int cp = (int)(idx & 255);
    size_t p = idx >> 8;
    int c = cp*2;
    if (c >= HID) {  // pad pair (510,511)
        *(v2h*)(g + p*512 + c) = (v2h){(_Float16)0.f, (_Float16)0.f};
        return;
    }
    int x = (int)(p & (WW-1));
    int y = (int)((p >> 8) & (HH-1));
    int b = (int)(p >> 16);
    float s1a = dwb[c], s1b = dwb[c+1];
    float s2a = dwb[c+HID], s2b = dwb[c+1+HID];
    for (int dy = -1; dy <= 1; ++dy) {
        int yy = y + dy; if (yy < 0 || yy >= HH) continue;
        for (int dx = -1; dx <= 1; ++dx) {
            int xx = x + dx; if (xx < 0 || xx >= WW) continue;
            size_t hb = (((size_t)(b*HH + yy))*WW + xx)*1024;
            int w = (dy+1)*3 + (dx+1);
            v2h h1 = *(const v2h*)(hbuf + hb + c);
            v2h h2 = *(const v2h*)(hbuf + hb + c + HID);
            float2 w1 = *(const float2*)(dwwT + w*1024 + c);
            float2 w2 = *(const float2*)(dwwT + w*1024 + c + HID);
            s1a += w1.x*(float)h1[0]; s1b += w1.y*(float)h1[1];
            s2a += w2.x*(float)h2[0]; s2b += w2.y*(float)h2[1];
        }
    }
    float ga = 0.5f*s1a*(1.f + erff(s1a*0.70710678118654752f));
    float gb = 0.5f*s1b*(1.f + erff(s1b*0.70710678118654752f));
    v2h r; r[0] = (_Float16)(ga*s2a); r[1] = (_Float16)(gb*s2b);
    *(v2h*)(g + p*512 + c) = r;
}

// pout: out(16px x 192) = g(16x512) @ pout -> outpre f16 NCHW
__global__ __launch_bounds__(128) void pout_gemm_kernel(const _Float16* __restrict__ g,
                                                        const _Float16* __restrict__ poutP,
                                                        const float*    __restrict__ poutb,
                                                        _Float16*       __restrict__ outpre) {
    int wave = blockIdx.x*(blockDim.x >> 5) + (threadIdx.x >> 5);
    int mt = wave / 3;
    int ng = wave % 3;
    size_t p0 = (size_t)mt * 16;
    const _Float16* A = g + p0*512;
    int n = threadIdx.x & 15, half = (threadIdx.x >> 4) & 1;
    v8f acc[4] = {};
    for (int ks = 0; ks < 16; ++ks) {
        v16h a = load_A16x32(A, 512, ks*32);
        if (ks < 15)
            __builtin_prefetch(poutP + ((size_t)((ng*4)*16 + ks + 1) << 9), 0, 3);
#pragma unroll
        for (int nt = 0; nt < 4; ++nt)
            acc[nt] = wmma_f16(a, load_Bfrag(poutP, (ng*4 + nt)*16 + ks), acc[nt]);
    }
#pragma unroll
    for (int nt = 0; nt < 4; ++nt) {
        int ch = ng*64 + nt*16 + n;
        float bias = poutb[ch];
#pragma unroll
        for (int v = 0; v < 8; ++v) {
            size_t p = p0 + v + 8*half;
            size_t bi = p >> 16, pix = p & 65535;
            outpre[(bi*DIMC + ch)*HW + pix] = (_Float16)(acc[nt][v] + bias);
        }
    }
}

// SE: per-(b,c) spatial mean of outpre (v8h vectorized)
__global__ __launch_bounds__(256) void chanmean_kernel(const _Float16* __restrict__ outpre,
                                                       float* __restrict__ chmean) {
    int bc = blockIdx.x;  // 0..767
    const v8h* p = (const v8h*)(outpre + (size_t)bc*HW);
    float s = 0.f;
    for (int i = threadIdx.x; i < HW/8; i += 256) {
        v8h v = p[i];
#pragma unroll
        for (int j = 0; j < 8; ++j) s += (float)v[j];
    }
    __shared__ float red[256];
    red[threadIdx.x] = s; __syncthreads();
    for (int o = 128; o > 0; o >>= 1) {
        if (threadIdx.x < o) red[threadIdx.x] += red[threadIdx.x + o];
        __syncthreads();
    }
    if (threadIdx.x == 0) chmean[bc] = red[0] / (float)HW;
}

// SE MLP: y = sigmoid(relu(mean @ w1) @ w2)
__global__ __launch_bounds__(768) void se_kernel(const float* __restrict__ chmean,
                                                 const float* __restrict__ w1,   // (192,24)
                                                 const float* __restrict__ w2,   // (24,192)
                                                 float* __restrict__ y) {
    __shared__ float r[BATCH][REDC];
    int t = threadIdx.x;
    if (t < BATCH*REDC) {
        int b = t / REDC, j = t % REDC;
        float a = 0.f;
        for (int c = 0; c < DIMC; ++c) a += chmean[b*DIMC + c]*w1[c*REDC + j];
        r[b][j] = fmaxf(a, 0.f);
    }
    __syncthreads();
    int b = t / DIMC, c = t % DIMC;
    float a = 0.f;
    for (int j = 0; j < REDC; ++j) a += r[b][j]*w2[j*DIMC + c];
    y[t] = 1.f/(1.f + expf(-a));
}

// out = x2 + outpre * y[b,c]   (pairwise vectorized)
__global__ void final_kernel(const float* __restrict__ x2, const _Float16* __restrict__ outpre,
                             const float* __restrict__ y, float* __restrict__ out) {
    size_t i = (size_t)blockIdx.x*blockDim.x + threadIdx.x;
    if (i >= (size_t)NPIX*DIMC/2) return;
    size_t i2 = i*2;
    size_t bc = i2 >> 16;
    float yv = y[bc];
    float2 xv = *(const float2*)(x2 + i2);
    v2h ov = *(const v2h*)(outpre + i2);
    float2 r;
    r.x = xv.x + (float)ov[0]*yv;
    r.y = xv.y + (float)ov[1]*yv;
    *(float2*)(out + i2) = r;
}

// ---------------- host launcher ----------------

extern "C" void kernel_launch(void* const* d_in, const int* in_sizes, int n_in,
                              void* d_out, int out_size, void* d_ws, size_t ws_size,
                              hipStream_t stream) {
    (void)in_sizes; (void)n_in; (void)out_size; (void)ws_size;
    const float* x      = (const float*)d_in[0];
    const float* temb   = (const float*)d_in[1];
    const float* gn1_g  = (const float*)d_in[2];
    const float* gn1_b  = (const float*)d_in[3];
    const float* aw1_w1 = (const float*)d_in[4];
    const float* aw1_b1 = (const float*)d_in[5];
    const float* aw1_w2 = (const float*)d_in[6];
    const float* aw1_b2 = (const float*)d_in[7];
    const float* qkv_w  = (const float*)d_in[8];
    const float* qkv_b  = (const float*)d_in[9];
    const float* proj_w = (const float*)d_in[10];
    const float* proj_b = (const float*)d_in[11];
    const float* gn2_g  = (const float*)d_in[12];
    const float* gn2_b  = (const float*)d_in[13];
    const float* aw2_w1 = (const float*)d_in[14];
    const float* aw2_b1 = (const float*)d_in[15];
    const float* aw2_w2 = (const float*)d_in[16];
    const float* aw2_b2 = (const float*)d_in[17];
    const float* pin_w  = (const float*)d_in[18];
    const float* pin_b  = (const float*)d_in[19];
    const float* dw_w   = (const float*)d_in[20];
    const float* dw_b   = (const float*)d_in[21];
    const float* pout_w = (const float*)d_in[22];
    const float* pout_b = (const float*)d_in[23];
    const float* se_w1  = (const float*)d_in[24];
    const float* se_w2  = (const float*)d_in[25];
    float* out = (float*)d_out;

    char* ws = (char*)d_ws;
    size_t off = 0;
    auto alloc = [&](size_t bytes) -> void* {
        void* p = ws + off;
        off = (off + bytes + 255) & ~(size_t)255;
        return p;
    };

    float*    stats1 = (float*)alloc(8*sizeof(float));
    float*    stats2 = (float*)alloc(8*sizeof(float));
    float*    s1     = (float*)alloc((size_t)BATCH*4*DIMC*4);
    float*    s2     = (float*)alloc((size_t)BATCH*4*DIMC*4);
    float*    chmean = (float*)alloc((size_t)BATCH*DIMC*4);
    float*    sey    = (float*)alloc((size_t)BATCH*DIMC*4);
    float*    dwwT   = (float*)alloc((size_t)9*1024*4);
    _Float16* qkvP   = (_Float16*)alloc((size_t)DIMC*3*DIMC*2);
    _Float16* projP  = (_Float16*)alloc((size_t)DIMC*DIMC*2);
    _Float16* pinP   = (_Float16*)alloc((size_t)DIMC*1024*2);
    _Float16* poutP  = (_Float16*)alloc((size_t)512*DIMC*2);
    _Float16* t1     = (_Float16*)alloc((size_t)NPIX*DIMC*2);   // 100.7 MB
    _Float16* t2     = (_Float16*)alloc((size_t)NPIX*DIMC*2);   // 100.7 MB
    float*    x2     = (float*)alloc((size_t)NPIX*DIMC*4);      // 201.3 MB
    _Float16* hbuf   = (_Float16*)alloc((size_t)NPIX*1024*2);   // 536.9 MB
    _Float16* gbuf   = (_Float16*)alloc((size_t)NPIX*512*2);    // 268.4 MB
    _Float16* outpre = (_Float16*)alloc((size_t)NPIX*DIMC*2);   // 100.7 MB

    hipMemsetAsync(stats1, 0, 8*sizeof(float), stream);
    hipMemsetAsync(stats2, 0, 8*sizeof(float), stream);

    // fragment-major f16 weight packing (deterministic every call)
    packB_kernel<<<(DIMC*3*DIMC + 255)/256, 256, 0, stream>>>(qkv_w,  qkvP, 192, 576, 192, 576, 576, 0);
    packB_kernel<<<(DIMC*DIMC   + 255)/256, 256, 0, stream>>>(proj_w, projP, 192, 192, 192, 192, 192, 0);
    packB_kernel<<<(DIMC*1024   + 255)/256, 256, 0, stream>>>(pin_w,  pinP, 192, 1024, 192, 1020, 192, 1);
    packB_kernel<<<(512*DIMC    + 255)/256, 256, 0, stream>>>(pout_w, poutP, 512, 192, 510, 192, 510, 1);
    dwwT_kernel <<<(9*1024 + 255)/256, 256, 0, stream>>>(dw_w, dwwT);

    // ---- stage 1: GN + adawm + window attention + residual ----
    moments_kernel<<<BATCH*1024, 256, 0, stream>>>(x, stats1, DIMC*HW, 1024);
    scale_mlp_kernel<<<BATCH, TIMED, 0, stream>>>(temb, aw1_w1, aw1_b1, aw1_w2, aw1_b2, s1);
    norm_adawm_kernel<<<(BATCH*(DIMC/2)*16384 + 255)/256, 256, 0, stream>>>(x, stats1, gn1_g, gn1_b, s1, t1);
    win_attn_kernel<<<BATCH*64*64, 32, 0, stream>>>(t1, x, qkvP, qkv_b, projP, proj_b, x2);

    // ---- stage 2: GN + adawm + GDFN + SE + residual ----
    moments_kernel<<<BATCH*1024, 256, 0, stream>>>(x2, stats2, DIMC*HW, 1024);
    scale_mlp_kernel<<<BATCH, TIMED, 0, stream>>>(temb, aw2_w1, aw2_b1, aw2_w2, aw2_b2, s2);
    norm_adawm_kernel<<<(BATCH*(DIMC/2)*16384 + 255)/256, 256, 0, stream>>>(x2, stats2, gn2_g, gn2_b, s2, t2);

    pin_gemm_kernel<<<65536, 128, 0, stream>>>(t2, pinP, pin_b, hbuf);
    dwgate_kernel<<<(int)(((size_t)NPIX*256 + 255)/256), 256, 0, stream>>>(hbuf, dwwT, dw_b, gbuf);
    pout_gemm_kernel<<<12288, 128, 0, stream>>>(gbuf, poutP, pout_b, outpre);
    chanmean_kernel<<<BATCH*DIMC, 256, 0, stream>>>(outpre, chmean);
    se_kernel<<<1, 768, 0, stream>>>(chmean, se_w1, se_w2, sey);
    final_kernel<<<(int)(((size_t)NPIX*DIMC/2 + 255)/256), 256, 0, stream>>>(x2, outpre, sey, out);
}
